// LocalContextAttentionBlock_42649025249953
// MI455X (gfx1250) — matrix operand order repaired and checked by reading, not compiled, not run on hardware
//
#include <hip/hip_runtime.h>
#include <hip/hip_bf16.h>

// ---------------------------------------------------------------------------
// LocalContextAttentionBlock for MI455X (gfx1250, wave32, WMMA bf16)
//   5x 1x1-conv+BN+ReLU (GEMM [256x256]x[256x65536], 43 of 48 GFLOP)
//       -> v_wmma_f32_16x16x32_bf16, 32x32 per-wave tile, 4 WMMA/K-step,
//          register-prefetch double buffering, BN stats fused in epilogue.
//   q/k final BN+ReLU also transposes to pixel-major [B][HW][C] bf16 so the
//   9x9 similarity dots read contiguous 512B vectors (b128 loads) instead of
//   32KB-strided 2B gathers. All gather tensors are L2-resident (192 MB L2).
// ---------------------------------------------------------------------------

typedef __attribute__((ext_vector_type(16))) __bf16 v16bf;
typedef __attribute__((ext_vector_type(8)))  float  v8f;

#define B_   4
#define C_   256
#define H_   128
#define W_   128
#define HW_  (H_ * W_)
#define KS_  9
#define KK_  (KS_ * KS_)
#define PAD_ 4
#define NE_   ((size_t)B_ * C_ * HW_)   // 16,777,216 elements
#define NCOL_ ((size_t)B_ * HW_)        // 65,536 pixels (GEMM N dimension)

// -------------------------------------------------------------- f32 -> bf16
__global__ __launch_bounds__(256)
void cvt_bf16(const float* __restrict__ in, __bf16* __restrict__ out, size_t n) {
  size_t i = (size_t)blockIdx.x * 256 + threadIdx.x;
  if (i < n) out[i] = (__bf16)in[i];
}

// ------------------------------------------------------------- stats helpers
__global__ __launch_bounds__(256)
void zero_stats(float* __restrict__ gsum, float* __restrict__ gsq) {
  gsum[threadIdx.x] = 0.f;
  gsq [threadIdx.x] = 0.f;
}

__global__ __launch_bounds__(256)
void finalize_stats(const float* __restrict__ gsum, const float* __restrict__ gsq,
                    float* __restrict__ mean, float* __restrict__ invstd) {
  const int c = threadIdx.x;
  const float inv_n = 1.0f / (float)(B_ * HW_);
  float m   = gsum[c] * inv_n;
  float var = gsq[c] * inv_n - m * m;
  mean[c]   = m;
  invstd[c] = rsqrtf(var + 1e-5f);
}

// ----------------------------------------------------------- WMMA bf16 GEMM
// Y[o, n] = sum_c Wt[o, c] * X[c, n], X laid out [B][C][HW], n = b*HW+hw.
// Block: 256 threads = 8 waves (2 M-groups x 4 N-groups), block tile 64x128.
// Each wave owns a 32x32 tile -> 2x2 fragments, 4 WMMA per 32-wide K-step,
// register-prefetch double buffering of the next K slab over the WMMAs.
// Epilogue fuses BN batch statistics (sum, sum of squares per channel).
__global__ __launch_bounds__(256)
void gemm_bf16_wmma(const __bf16* __restrict__ Wt,   // [C_ x C_] row-major (o,c)
                    const __bf16* __restrict__ X,    // [B][C][HW] bf16
                    float* __restrict__ Y,           // [B][C][HW] f32
                    float* __restrict__ gsum,        // [C_] global sum
                    float* __restrict__ gsq)         // [C_] global sum of squares
{
  __shared__ __bf16 As [64][40];    // [m][k], +8 pad (rows 80B, 16B aligned)
  __shared__ __bf16 Bst[128][40];   // [n][k] transposed, contiguous-K reads
  __shared__ float  csum[64], csq[64];

  const int tid  = threadIdx.x;
  const int lane = tid & 31;
  const int wave = tid >> 5;
  const int wmB  = (wave >> 2) * 32;   // 0 / 32        (M sub-tile base)
  const int wnB  = (wave & 3)  * 32;   // 0/32/64/96    (N sub-tile base)

  const int om_base = blockIdx.x * 64;
  const int on_base = blockIdx.y * 128;     // 128 | HW_, one batch per block
  const int b   = on_base / HW_;
  const int hw0 = on_base % HW_;
  const __bf16* __restrict__ Xb = X + (size_t)b * C_ * HW_ + hw0;

  if (tid < 64) { csum[tid] = 0.f; csq[tid] = 0.f; }

  // A staging: 64x32 elems, 8 contiguous K per thread (16B global loads)
  const int am = tid >> 2;             // 0..63
  const int ak = (tid & 3) * 8;        // 0..24
  // B staging: 32x128 elems as two 8-wide N chunks per thread (16B loads)
  const int c0  = tid * 2, c1 = c0 + 1;
  const int bk0 = c0 >> 4, bn0 = (c0 & 15) * 8;
  const int bk1 = c1 >> 4, bn1 = (c1 & 15) * 8;

  const int ml  = lane & 15;
  const int kA  = (lane < 16) ? 0 : 16;
  const int kB  = (lane < 16) ? 0 : 16;
  const int hi8 = (lane < 16) ? 0 : 8;

  v8f acc00 = {0.f,0.f,0.f,0.f,0.f,0.f,0.f,0.f};
  v8f acc01 = acc00, acc10 = acc00, acc11 = acc00;

  __bf16 ra[8], rb0[8], rb1[8];

  { // prologue: fetch K slab 0 into registers
    const __bf16* sA = Wt + (size_t)(om_base + am) * C_ + ak;
    const __bf16* s0 = Xb + (size_t)bk0 * HW_ + bn0;
    const __bf16* s1 = Xb + (size_t)bk1 * HW_ + bn1;
    #pragma unroll
    for (int j = 0; j < 8; ++j) { ra[j] = sA[j]; rb0[j] = s0[j]; rb1[j] = s1[j]; }
  }

  for (int kk = 0; kk < C_; kk += 32) {
    // registers -> LDS
    #pragma unroll
    for (int j = 0; j < 8; ++j) As[am][ak + j] = ra[j];
    #pragma unroll
    for (int j = 0; j < 8; ++j) { Bst[bn0 + j][bk0] = rb0[j]; Bst[bn1 + j][bk1] = rb1[j]; }
    __syncthreads();

    if (kk + 32 < C_) { // prefetch next K slab; overlaps the WMMA sequence
      const int kn = kk + 32;
      const __bf16* sA = Wt + (size_t)(om_base + am) * C_ + kn + ak;
      const __bf16* s0 = Xb + (size_t)(kn + bk0) * HW_ + bn0;
      const __bf16* s1 = Xb + (size_t)(kn + bk1) * HW_ + bn1;
      #pragma unroll
      for (int j = 0; j < 8; ++j) { ra[j] = sA[j]; rb0[j] = s0[j]; rb1[j] = s1[j]; }
    }

    // build fragments (contiguous LDS rows -> ds_load_b128 pairs)
    const __bf16* pa0 = &As [wmB +      ml][kA];
    const __bf16* pa1 = &As [wmB + 16 + ml][kA];
    const __bf16* pb0 = &Bst[wnB +      ml][kB];
    const __bf16* pb1 = &Bst[wnB + 16 + ml][kB];
    v16bf a0, a1, b0v, b1v;
    #pragma unroll
    for (int i = 0; i < 8; ++i) {
      a0[i] = pa0[i]; a0[i + 8] = pa0[i + 8];
      a1[i] = pa1[i]; a1[i + 8] = pa1[i + 8];
    }
    #pragma unroll
    for (int i = 0; i < 16; ++i) { b0v[i] = pb0[i]; b1v[i] = pb1[i]; }

    acc00 = __builtin_amdgcn_wmma_f32_16x16x32_bf16(false, a0, false, b0v, (short)0, acc00, false, false);
    acc01 = __builtin_amdgcn_wmma_f32_16x16x32_bf16(false, a0, false, b1v, (short)0, acc01, false, false);
    acc10 = __builtin_amdgcn_wmma_f32_16x16x32_bf16(false, a1, false, b0v, (short)0, acc10, false, false);
    acc11 = __builtin_amdgcn_wmma_f32_16x16x32_bf16(false, a1, false, b1v, (short)0, acc11, false, false);
    __syncthreads();
  }

  // ---- write D tiles: lane n = lane&15; VGPR r -> M = r + (lane<16?0:8) ----
  const int col = hw0 + wnB + ml;
  float* dstB = Y + (size_t)b * C_ * HW_;
  #pragma unroll
  for (int r = 0; r < 8; ++r) {
    const int ch0 = om_base + wmB + r + hi8;
    const int ch1 = ch0 + 16;
    dstB[(size_t)ch0 * HW_ + col     ] = acc00[r];
    dstB[(size_t)ch0 * HW_ + col + 16] = acc01[r];
    dstB[(size_t)ch1 * HW_ + col     ] = acc10[r];
    dstB[(size_t)ch1 * HW_ + col + 16] = acc11[r];
  }

  // ---- fused BN statistics: reduce this wave's 32 columns per channel ----
  #pragma unroll
  for (int mi = 0; mi < 2; ++mi) {
    #pragma unroll
    for (int r = 0; r < 8; ++r) {
      const float v0 = (mi == 0) ? acc00[r] : acc10[r];
      const float v1 = (mi == 0) ? acc01[r] : acc11[r];
      float s  = v0 + v1;
      float s2 = v0 * v0 + v1 * v1;
      #pragma unroll
      for (int m = 1; m < 16; m <<= 1) {   // reduce across the 16-lane N group
        s  += __shfl_xor(s,  m, 32);
        s2 += __shfl_xor(s2, m, 32);
      }
      if ((lane & 15) == 0) {
        const int chl = wmB + mi * 16 + r + hi8;   // local channel 0..63
        atomicAdd(&csum[chl], s);
        atomicAdd(&csq [chl], s2);
      }
    }
  }
  __syncthreads();
  if (tid < 64) {
    atomicAdd(&gsum[om_base + tid], csum[tid]);
    atomicAdd(&gsq [om_base + tid], csq [tid]);
  }
}

// --------------------------------------------------- fused BN + ReLU + cast
__global__ __launch_bounds__(256)
void bn_relu(const float* __restrict__ Y, const float* __restrict__ mean,
             const float* __restrict__ invstd, const float* __restrict__ gamma,
             const float* __restrict__ beta,
             __bf16* __restrict__ out_bf, float* __restrict__ out_f32) {
  size_t i = (size_t)blockIdx.x * 256 + threadIdx.x;
  if (i >= NE_) return;
  const int c = (int)((i / HW_) % C_);
  float v = (Y[i] - mean[c]) * invstd[c] * gamma[c] + beta[c];
  v = fmaxf(v, 0.f);
  if (out_bf)  out_bf[i]  = (__bf16)v;
  if (out_f32) out_f32[i] = v;
}

// ------------------- BN + ReLU + transpose to pixel-major [B][HW][C] bf16
// 32x32 (channel x pixel) tile via LDS: reads coalesced along HW, writes
// coalesced along C. Gives sim_softmax contiguous 512B per-pixel vectors.
__global__ __launch_bounds__(256)
void bn_relu_tr(const float* __restrict__ Y, const float* __restrict__ mean,
                const float* __restrict__ invstd, const float* __restrict__ gamma,
                const float* __restrict__ beta, __bf16* __restrict__ outT) {
  __shared__ float tile[32][33];
  const int tx = threadIdx.x & 31;       // pixel (read) / channel (write)
  const int ty = threadIdx.x >> 5;       // 0..7
  const int c0 = blockIdx.x * 32;
  const int n0 = blockIdx.y * 32;        // global pixel base (32 | HW_)
  const int b  = n0 / HW_;
  const int hw0 = n0 % HW_;

  #pragma unroll
  for (int j = 0; j < 4; ++j) {
    const int ci = ty + j * 8;           // 0..31
    const int c  = c0 + ci;
    float v = Y[((size_t)b * C_ + c) * HW_ + hw0 + tx];
    v = (v - mean[c]) * invstd[c] * gamma[c] + beta[c];
    tile[ci][tx] = fmaxf(v, 0.f);
  }
  __syncthreads();
  #pragma unroll
  for (int j = 0; j < 4; ++j) {
    const int pi = ty + j * 8;           // pixel within tile
    outT[(size_t)(n0 + pi) * C_ + c0 + tx] = (__bf16)tile[tx][pi];
  }
}

// --------------------- per-pixel 9x9 similarity + softmax (contiguous dots)
// QT/KT are pixel-major: 256 bf16 (512B, 512B-aligned) per pixel.
__global__ __launch_bounds__(96)
void sim_softmax(const __bf16* __restrict__ QT, const __bf16* __restrict__ KT,
                 float* __restrict__ Wgt) {
  __shared__ float qs[C_];
  __shared__ float simv[KK_];
  const int pix = blockIdx.x;            // 0..65535
  const int b  = pix / HW_;
  const int hw = pix % HW_;
  const int h  = hw / W_;
  const int w  = hw % W_;

  const __bf16* qp = QT + (size_t)pix * C_;
  for (int c = threadIdx.x; c < C_; c += 96) qs[c] = (float)qp[c];
  __syncthreads();

  const int t = threadIdx.x;
  if (t < KK_) {
    const int di = t / KS_, dj = t % KS_;
    const int hh = h + di - PAD_, ww = w + dj - PAD_;
    float s = 0.f;
    if (hh >= 0 && hh < H_ && ww >= 0 && ww < W_) {
      const __bf16* kp = KT + ((size_t)b * HW_ + hh * W_ + ww) * C_;
      #pragma unroll 8
      for (int c = 0; c < C_; ++c) s += qs[c] * (float)kp[c];
    }
    simv[t] = s * 0.0625f;               // 1/sqrt(C)
  }
  __syncthreads();
  if (t < KK_) {
    float mx = -1e30f;
    for (int j = 0; j < KK_; ++j) mx = fmaxf(mx, simv[j]);
    float se = 0.f;
    for (int j = 0; j < KK_; ++j) se += __expf(simv[j] - mx);
    Wgt[(size_t)pix * KK_ + t] = __expf(simv[t] - mx) / se;
  }
}

// ------------------------------------------ local weighted aggregation of v
__global__ __launch_bounds__(256)
void weighting(const float* __restrict__ V, const float* __restrict__ Wgt,
               float* __restrict__ Out) {
  size_t i = (size_t)blockIdx.x * 256 + threadIdx.x;
  if (i >= NE_) return;
  const int hw = (int)(i % HW_);
  const int c  = (int)((i / HW_) % C_);
  const int b  = (int)(i / ((size_t)C_ * HW_));
  const int h = hw / W_, w = hw % W_;
  const float* wp = Wgt + ((size_t)b * HW_ + hw) * KK_;
  const float* vb = V + ((size_t)b * C_ + c) * HW_;
  float acc = 0.f;
  #pragma unroll
  for (int di = 0; di < KS_; ++di) {
    const int hh = h + di - PAD_;
    if (hh < 0 || hh >= H_) continue;
    #pragma unroll
    for (int dj = 0; dj < KS_; ++dj) {
      const int ww = w + dj - PAD_;
      if (ww < 0 || ww >= W_) continue;
      acc += wp[di * KS_ + dj] * vb[hh * W_ + ww];
    }
  }
  Out[i] = acc;
}

// ---------------------------------------------------------------------------
extern "C" void kernel_launch(void* const* d_in, const int* in_sizes, int n_in,
                              void* d_out, int out_size, void* d_ws, size_t ws_size,
                              hipStream_t stream) {
  const float* target = (const float*)d_in[0];
  const float* source = (const float*)d_in[1];
  const float* qw1 = (const float*)d_in[2];
  const float* qg1 = (const float*)d_in[3];
  const float* qb1 = (const float*)d_in[4];
  const float* qw2 = (const float*)d_in[5];
  const float* qg2 = (const float*)d_in[6];
  const float* qb2 = (const float*)d_in[7];
  const float* kw1 = (const float*)d_in[8];
  const float* kg1 = (const float*)d_in[9];
  const float* kb1 = (const float*)d_in[10];
  const float* kw2 = (const float*)d_in[11];
  const float* kg2 = (const float*)d_in[12];
  const float* kb2 = (const float*)d_in[13];
  const float* vw  = (const float*)d_in[14];
  const float* vg  = (const float*)d_in[15];
  const float* vb  = (const float*)d_in[16];
  // d_in[17] = kernel_size (== 9, compiled in)

  // ---- carve workspace (~324 MB total) ----
  char* ws = (char*)d_ws;
  size_t off = 0;
  auto carve = [&](size_t bytes) -> void* {
    void* p = ws + off;
    off = (off + bytes + 255) & ~(size_t)255;
    return p;
  };
  __bf16* t_bf = (__bf16*)carve(NE_ * 2);
  __bf16* s_bf = (__bf16*)carve(NE_ * 2);
  __bf16* a_bf = (__bf16*)carve(NE_ * 2);   // intermediate activation (reused)
  __bf16* q_t  = (__bf16*)carve(NE_ * 2);   // q, pixel-major [B][HW][C]
  __bf16* k_t  = (__bf16*)carve(NE_ * 2);   // k, pixel-major [B][HW][C]
  __bf16* wq1b = (__bf16*)carve((size_t)C_ * C_ * 2);
  __bf16* wq2b = (__bf16*)carve((size_t)C_ * C_ * 2);
  __bf16* wk1b = (__bf16*)carve((size_t)C_ * C_ * 2);
  __bf16* wk2b = (__bf16*)carve((size_t)C_ * C_ * 2);
  __bf16* wvb  = (__bf16*)carve((size_t)C_ * C_ * 2);
  float*  y    = (float*)carve(NE_ * 4);    // raw GEMM output (reused)
  float*  vf   = (float*)carve(NE_ * 4);    // v activation (f32, channel-major)
  float*  wgt  = (float*)carve(NCOL_ * KK_ * 4);
  float*  gsum = (float*)carve(C_ * 4);
  float*  gsq  = (float*)carve(C_ * 4);
  float*  mean = (float*)carve(C_ * 4);
  float*  istd = (float*)carve(C_ * 4);

  const int BIG = (int)((NE_ + 255) / 256);            // 65536 blocks
  const int WSM = (int)(((size_t)C_ * C_ + 255) / 256);
  dim3 ggrid(C_ / 64, (unsigned)(NCOL_ / 128));        // (4, 512)
  dim3 trgrid(C_ / 32, (unsigned)(NCOL_ / 32));        // (8, 2048)

  // ---- downcast inputs & weights to bf16 ----
  cvt_bf16<<<BIG, 256, 0, stream>>>(target, t_bf, NE_);
  cvt_bf16<<<BIG, 256, 0, stream>>>(source, s_bf, NE_);
  cvt_bf16<<<WSM, 256, 0, stream>>>(qw1, wq1b, (size_t)C_ * C_);
  cvt_bf16<<<WSM, 256, 0, stream>>>(qw2, wq2b, (size_t)C_ * C_);
  cvt_bf16<<<WSM, 256, 0, stream>>>(kw1, wk1b, (size_t)C_ * C_);
  cvt_bf16<<<WSM, 256, 0, stream>>>(kw2, wk2b, (size_t)C_ * C_);
  cvt_bf16<<<WSM, 256, 0, stream>>>(vw,  wvb,  (size_t)C_ * C_);

  // ---- q path ----
  zero_stats<<<1, 256, 0, stream>>>(gsum, gsq);
  gemm_bf16_wmma<<<ggrid, 256, 0, stream>>>(wq1b, t_bf, y, gsum, gsq);
  finalize_stats<<<1, 256, 0, stream>>>(gsum, gsq, mean, istd);
  bn_relu<<<BIG, 256, 0, stream>>>(y, mean, istd, qg1, qb1, a_bf, nullptr);
  zero_stats<<<1, 256, 0, stream>>>(gsum, gsq);
  gemm_bf16_wmma<<<ggrid, 256, 0, stream>>>(wq2b, a_bf, y, gsum, gsq);
  finalize_stats<<<1, 256, 0, stream>>>(gsum, gsq, mean, istd);
  bn_relu_tr<<<trgrid, 256, 0, stream>>>(y, mean, istd, qg2, qb2, q_t);

  // ---- k path ----
  zero_stats<<<1, 256, 0, stream>>>(gsum, gsq);
  gemm_bf16_wmma<<<ggrid, 256, 0, stream>>>(wk1b, s_bf, y, gsum, gsq);
  finalize_stats<<<1, 256, 0, stream>>>(gsum, gsq, mean, istd);
  bn_relu<<<BIG, 256, 0, stream>>>(y, mean, istd, kg1, kb1, a_bf, nullptr);
  zero_stats<<<1, 256, 0, stream>>>(gsum, gsq);
  gemm_bf16_wmma<<<ggrid, 256, 0, stream>>>(wk2b, a_bf, y, gsum, gsq);
  finalize_stats<<<1, 256, 0, stream>>>(gsum, gsq, mean, istd);
  bn_relu_tr<<<trgrid, 256, 0, stream>>>(y, mean, istd, kg2, kb2, k_t);

  // ---- v path ----
  zero_stats<<<1, 256, 0, stream>>>(gsum, gsq);
  gemm_bf16_wmma<<<ggrid, 256, 0, stream>>>(wvb, s_bf, y, gsum, gsq);
  finalize_stats<<<1, 256, 0, stream>>>(gsum, gsq, mean, istd);
  bn_relu<<<BIG, 256, 0, stream>>>(y, mean, istd, vg, vb, nullptr, vf);

  // ---- local attention ----
  sim_softmax<<<(unsigned)NCOL_, 96, 0, stream>>>(q_t, k_t, wgt);
  weighting<<<BIG, 256, 0, stream>>>(vf, wgt, (float*)d_out);

  (void)in_sizes; (void)n_in; (void)out_size; (void)ws_size;
}
